// DMPNN_Change_678604832935
// MI455X (gfx1250) — matrix operand order repaired and checked by reading, not compiled, not run on hardware
//
#include <hip/hip_runtime.h>
#include <hip/hip_bf16.h>

typedef __bf16 bf16_t;
typedef __attribute__((ext_vector_type(16))) __bf16 v16bf;
typedef __attribute__((ext_vector_type(8)))  __bf16 v8bf;
typedef __attribute__((ext_vector_type(8)))  float  v8f;

#define NNODE 30000
#define NEDGE 480000
#define DDIM  256
#define FE    14
#define FIN   25
#define NGR   512
#define K1PAD 288   // 256 + 14 padded to 9*32
#define KGATE 768   // [h || r || h] -> Wih;Whh combined

#define EDGE_BLOCKS 512
#define NODE_BLOCKS 128

__device__ __forceinline__ bf16_t f2bf(float f) {
  union { float f; unsigned u; } a; a.f = f;
  unsigned r = a.u + (0x7FFFu + ((a.u >> 16) & 1u));   // RNE
  union { unsigned short s; bf16_t b; } o; o.s = (unsigned short)(r >> 16);
  return o.b;
}

__device__ __forceinline__ v8f vzero8() {
  v8f z;
#pragma unroll
  for (int i = 0; i < 8; i++) z[i] = 0.f;
  return z;
}

__device__ __forceinline__ v8f wmma_bf(v16bf a, v16bf b, v8f c) {
  return __builtin_amdgcn_wmma_f32_16x16x32_bf16(false, a, false, b, (short)0, c,
                                                 false, false);
}

// A-fragment (16x32 bf16, MxK) per ISA layout:
// lane = hf*16 + m ; elems 0..7 = K kbase+hf*8.. ; elems 8..15 = K kbase+16+hf*8..
template <typename P>
__device__ __forceinline__ v16bf load_a_frag(P rowA, int kbase, int hf) {
  union { v16bf v; v8bf h[2]; } u;
  u.h[0] = *(const v8bf*)(rowA + kbase + hf * 8);
  u.h[1] = *(const v8bf*)(rowA + kbase + 16 + hf * 8);
  return u.v;
}

// B-fragment (32x16 bf16, KxN) from transposed weights WT[N][Kstride]:
// lane = hf*16 + n ; elems 0..7 = K kbase+hf*8.. ; elems 8..15 = K kbase+16+hf*8..
template <typename P>
__device__ __forceinline__ v16bf load_b_frag(P WT, int col, int kstride, int kbase,
                                             int hf) {
  union { v16bf v; v8bf h[2]; } u;
  u.h[0] = *(const v8bf*)(WT + (long)col * kstride + kbase + hf * 8);
  u.h[1] = *(const v8bf*)(WT + (long)col * kstride + kbase + 16 + hf * 8);
  return u.v;
}

// order-preserving float<->uint for atomicMax-based segment max
__device__ __forceinline__ unsigned f_enc(float f) {
  unsigned u = __float_as_uint(f);
  return (u & 0x80000000u) ? ~u : (u | 0x80000000u);
}
__device__ __forceinline__ float f_dec(unsigned u) {
  unsigned b = (u & 0x80000000u) ? (u & 0x7FFFFFFFu) : ~u;
  return __uint_as_float(b);
}

// ---------- weight prep: bf16 transposed (N-major) copies ----------
__global__ __launch_bounds__(256) void k_prep(
    const float* __restrict__ Wm1, const float* __restrict__ Wm2,
    const float* __restrict__ Wr,  const float* __restrict__ Wih,
    const float* __restrict__ Whh,
    bf16_t* __restrict__ Wm1T, bf16_t* __restrict__ Wm2T,
    bf16_t* __restrict__ WrT,  bf16_t* __restrict__ WgT) {
  int i = blockIdx.x * 256 + threadIdx.x;
  if (i < DDIM * K1PAD) {            // Wm1T[256][288], K padded with zeros
    int n = i / K1PAD, k = i % K1PAD;
    Wm1T[i] = f2bf(k < (DDIM + FE) ? Wm1[k * DDIM + n] : 0.f);
  }
  if (i < DDIM * DDIM) {             // Wm2T / WrT [256][256]
    int n = i / DDIM, k = i % DDIM;
    Wm2T[i] = f2bf(Wm2[k * DDIM + n]);
    WrT[i]  = f2bf(Wr[k * DDIM + n]);
  }
  if (i < 4 * DDIM * KGATE) {        // WgT[1024][768] = [Wih ; Whh] transposed
    int n = i / KGATE, k = i % KGATE;
    float v = (k < 2 * DDIM) ? Wih[k * 4 * DDIM + n] : Whh[(k - 2 * DDIM) * 4 * DDIM + n];
    WgT[i] = f2bf(v);
  }
}

// ---------- h0 = relu(x@W0+b0) -> bf16 ; also zero agg ----------
__global__ __launch_bounds__(256) void k_h0(
    const float* __restrict__ x, const float* __restrict__ W0,
    const float* __restrict__ b0, bf16_t* __restrict__ h0b,
    float* __restrict__ agg) {
  __shared__ float xs[FIN];
  int n = blockIdx.x, d = threadIdx.x;
  if (d < FIN) xs[d] = x[n * FIN + d];
  __syncthreads();
  float s = b0[d];
#pragma unroll
  for (int k = 0; k < FIN; k++) s = fmaf(xs[k], W0[k * DDIM + d], s);
  s = fmaxf(s, 0.f);
  h0b[(long)n * DDIM + d] = f2bf(s);
  agg[(long)n * DDIM + d] = 0.f;
}

// ---- layer-2 pass for one 16-edge half: relu+bias -> LDS t tile, GEMM vs Wm2T,
//      scatter-add into agg[dst] ----
__device__ __forceinline__ void layer2_half(
    const v8f* acc1, bf16_t* tb, const bf16_t* B2,
    const float* __restrict__ bm1, const float* __restrict__ bm2,
    const int* __restrict__ dst, float* __restrict__ agg,
    int ebase, int m, int hf) {
  // bias + relu -> per-wave LDS tile (row-major [16 edges][256]) in bf16
#pragma unroll
  for (int t = 0; t < 16; t++) {
    int col = t * 16 + m;
    float bv = bm1[col];
#pragma unroll
    for (int j = 0; j < 8; j++) {
      int row = j + 8 * hf;                    // C/D layout: M = j + 8*half
      float v = acc1[t][j] + bv;
      tb[row * DDIM + col] = f2bf(v > 0.f ? v : 0.f);
    }
  }
  v8f a2[16];
#pragma unroll
  for (int t = 0; t < 16; t++) a2[t] = vzero8();
  const bf16_t* rowT = tb + m * DDIM;
#pragma unroll
  for (int c = 0; c < 8; c++) {
    v16bf af2 = load_a_frag(rowT, c * 32, hf);
#pragma unroll
    for (int g = 0; g < 4; g++) {
      v16bf bf[4];
#pragma unroll
      for (int q = 0; q < 4; q++)
        bf[q] = load_b_frag(B2, (g * 4 + q) * 16 + m, DDIM, c * 32, hf);
#pragma unroll
      for (int q = 0; q < 4; q++) a2[g * 4 + q] = wmma_bf(af2, bf[q], a2[g * 4 + q]);
    }
  }
  int drow[8];
#pragma unroll
  for (int j = 0; j < 8; j++) drow[j] = dst[ebase + 8 * hf + j];
#pragma unroll
  for (int t = 0; t < 16; t++) {
    int col = t * 16 + m;
    float bv = bm2[col];
#pragma unroll
    for (int j = 0; j < 8; j++)
      atomicAdd(agg + (long)drow[j] * DDIM + col, a2[t][j] + bv);
  }
}

// ---------- fused edge MLP, weight-stationary in LDS, M=32 blocking ----------
// LDS: Wm1T [256][288] (144KB) + Wm2T [256][256] (128KB) + per-wave t tile (32KB)
__global__ __launch_bounds__(128) void k_edges(
    const bf16_t* __restrict__ h0b, const float* __restrict__ ea,
    const bf16_t* __restrict__ Wm1T, const bf16_t* __restrict__ Wm2T,
    const float* __restrict__ bm1, const float* __restrict__ bm2,
    const int* __restrict__ src, const int* __restrict__ dst,
    float* __restrict__ agg) {
  extern __shared__ __align__(16) char smem[];
  bf16_t* B1 = (bf16_t*)smem;                 // [256][288]
  bf16_t* B2 = B1 + DDIM * K1PAD;             // [256][256]
  bf16_t* TB = B2 + DDIM * DDIM;              // [4][16*256]

  // cooperative stage of both weight matrices into LDS (once per block)
  {
    const uint4* g1 = (const uint4*)Wm1T;
    uint4* l1 = (uint4*)B1;
    for (int i = threadIdx.x; i < DDIM * K1PAD / 8; i += 128) l1[i] = g1[i];
    const uint4* g2 = (const uint4*)Wm2T;
    uint4* l2 = (uint4*)B2;
    for (int i = threadIdx.x; i < DDIM * DDIM / 8; i += 128) l2[i] = g2[i];
  }
  __syncthreads();

  const int lane = threadIdx.x & 31;
  const int wv = threadIdx.x >> 5;
  const int m = lane & 15;
  const int hf = lane >> 4;
  bf16_t* tb = TB + wv * 16 * DDIM;
  const int wave0 = blockIdx.x * 4 + wv;
  const int nwaves = EDGE_BLOCKS * 4;

  for (int tile = wave0; tile < NEDGE / 32; tile += nwaves) {
    const int e0 = tile * 32;
    const int se0 = src[e0 + m];
    const int se1 = src[e0 + 16 + m];
    const bf16_t* rowA0 = h0b + (long)se0 * DDIM;
    const bf16_t* rowA1 = h0b + (long)se1 * DDIM;

    // edge-feature A fragments for both halves (K chunk 8: feats 0..13, rest 0)
    v16bf afe0, afe1;
    {
      const float* er0 = ea + (long)(e0 + m) * FE;
      const float* er1 = ea + (long)(e0 + 16 + m) * FE;
      union { v16bf v; v8bf h[2]; } u0, u1;
#pragma unroll
      for (int i = 0; i < 8; i++) {
        int f = hf * 8 + i;
        u0.h[0][i] = f2bf(f < FE ? er0[f] : 0.f);
        u1.h[0][i] = f2bf(f < FE ? er1[f] : 0.f);
        u0.h[1][i] = f2bf(0.f);
        u1.h[1][i] = f2bf(0.f);
      }
      afe0 = u0.v; afe1 = u1.v;
    }

    v8f acc0[16], acc1[16];
#pragma unroll
    for (int t = 0; t < 16; t++) { acc0[t] = vzero8(); acc1[t] = vzero8(); }

    // layer 1: K = 9 chunks; each LDS B-fragment feeds two WMMAs (M=32 blocking)
#pragma unroll
    for (int c = 0; c < 9; c++) {
      v16bf a0 = (c < 8) ? load_a_frag(rowA0, c * 32, hf) : afe0;
      v16bf a1 = (c < 8) ? load_a_frag(rowA1, c * 32, hf) : afe1;
#pragma unroll
      for (int g = 0; g < 4; g++) {
        v16bf bf[4];
#pragma unroll
        for (int q = 0; q < 4; q++)
          bf[q] = load_b_frag(B1, (g * 4 + q) * 16 + m, K1PAD, c * 32, hf);
#pragma unroll
        for (int q = 0; q < 4; q++) {
          acc0[g * 4 + q] = wmma_bf(a0, bf[q], acc0[g * 4 + q]);
          acc1[g * 4 + q] = wmma_bf(a1, bf[q], acc1[g * 4 + q]);
        }
      }
    }

    // layer 2 + scatter, one 16-edge half at a time (keeps t tile at 8KB/wave;
    // half0's accumulators die before half1's layer-2 registers go live)
    layer2_half(acc0, tb, B2, bm1, bm2, dst, agg, e0, m, hf);
    layer2_half(acc1, tb, B2, bm1, bm2, dst, agg, e0 + 16, m, hf);
  }
}

// ---------- out = relu(h0@Wr + br + agg), WrT stationary in LDS ----------
__global__ __launch_bounds__(128) void k_node_out(
    const bf16_t* __restrict__ h0b, const bf16_t* __restrict__ WrT,
    const float* __restrict__ br, const float* __restrict__ agg,
    float* __restrict__ outf) {
  extern __shared__ __align__(16) char smem[];
  bf16_t* BW = (bf16_t*)smem;                 // [256][256]
  {
    const uint4* g = (const uint4*)WrT;
    uint4* l = (uint4*)BW;
    for (int i = threadIdx.x; i < DDIM * DDIM / 8; i += 128) l[i] = g[i];
  }
  __syncthreads();

  const int lane = threadIdx.x & 31;
  const int wv = threadIdx.x >> 5;
  const int m = lane & 15, hf = lane >> 4;
  const int wave0 = blockIdx.x * 4 + wv;
  const int nwaves = NODE_BLOCKS * 4;

  for (int tile = wave0; tile < NNODE / 16; tile += nwaves) {
    const int n0 = tile * 16;
    const bf16_t* rowA = h0b + (long)(n0 + m) * DDIM;
    v16bf af[8];
#pragma unroll
    for (int c = 0; c < 8; c++) af[c] = load_a_frag(rowA, c * 32, hf);

    v8f acc[16];
#pragma unroll
    for (int t = 0; t < 16; t++) acc[t] = vzero8();
#pragma unroll
    for (int c = 0; c < 8; c++) {
#pragma unroll
      for (int g = 0; g < 4; g++) {
        v16bf bf[4];
#pragma unroll
        for (int q = 0; q < 4; q++)
          bf[q] = load_b_frag(BW, (g * 4 + q) * 16 + m, DDIM, c * 32, hf);
#pragma unroll
        for (int q = 0; q < 4; q++)
          acc[g * 4 + q] = wmma_bf(af[c], bf[q], acc[g * 4 + q]);
      }
    }
#pragma unroll
    for (int t = 0; t < 16; t++) {
      int col = t * 16 + m;
      float bv = br[col];
#pragma unroll
      for (int j = 0; j < 8; j++) {
        long idx = (long)(n0 + j + 8 * hf) * DDIM + col;
        float v = acc[t][j] + bv + agg[idx];
        outf[idx] = v > 0.f ? v : 0.f;
      }
    }
  }
}

// ---------- Set2Set ----------
__global__ __launch_bounds__(256) void k_init_state(float* hS, float* cS, float* rS) {
  int i = blockIdx.x * 256 + threadIdx.x;
  hS[i] = 0.f; cS[i] = 0.f; rS[i] = 0.f;
}

// qh[b] = [h(256) || r(256) || h(256)] in bf16  (gates = q_star@Wih + h@Whh)
__global__ __launch_bounds__(256) void k_qh(const float* __restrict__ hS,
                                            const float* __restrict__ rS,
                                            bf16_t* __restrict__ qhb) {
  int i = blockIdx.x * 256 + threadIdx.x;   // 512*768
  int b = i / KGATE, k = i % KGATE;
  float v = (k < DDIM) ? hS[b * DDIM + k]
          : (k < 2 * DDIM) ? rS[b * DDIM + k - DDIM]
          : hS[b * DDIM + k - 2 * DDIM];
  qhb[i] = f2bf(v);
}

// gates[512,1024] = qh[512,768] @ WgT' + bl  (WMMA, 128 waves)
__global__ __launch_bounds__(128) void k_gates(
    const bf16_t* __restrict__ qhb, const bf16_t* __restrict__ WgT,
    const float* __restrict__ bl, float* __restrict__ gates) {
  const int lane = threadIdx.x & 31;
  const int wave = blockIdx.x * 4 + (threadIdx.x >> 5);  // 0..127
  const int r0 = (wave & 31) * 16;                        // 32 row tiles
  const int c0 = (wave >> 5) * 256;                       // 4 col groups
  const int m = lane & 15, hf = lane >> 4;
  const bf16_t* rowA = qhb + (long)(r0 + m) * KGATE;

  v8f acc[16];
#pragma unroll
  for (int t = 0; t < 16; t++) acc[t] = vzero8();
  for (int c = 0; c < KGATE / 32; c++) {
    v16bf af = load_a_frag(rowA, c * 32, hf);
#pragma unroll
    for (int g = 0; g < 2; g++) {
      v16bf bf[8];
#pragma unroll
      for (int q = 0; q < 8; q++)
        bf[q] = load_b_frag(WgT, c0 + (g * 8 + q) * 16 + m, KGATE, c * 32, hf);
#pragma unroll
      for (int q = 0; q < 8; q++)
        acc[g * 8 + q] = wmma_bf(af, bf[q], acc[g * 8 + q]);
    }
  }
#pragma unroll
  for (int t = 0; t < 16; t++) {
    int col = c0 + t * 16 + m;
    float bv = bl[col];
#pragma unroll
    for (int j = 0; j < 8; j++)
      gates[(long)(r0 + j + 8 * hf) * 4 * DDIM + col] = acc[t][j] + bv;
  }
}

__global__ __launch_bounds__(256) void k_lstm(const float* __restrict__ gates,
                                              float* __restrict__ cS,
                                              float* __restrict__ hS) {
  int b = blockIdx.x, d = threadIdx.x;
  const float* g = gates + (long)b * 4 * DDIM;
  float ii = 1.f / (1.f + expf(-g[d]));
  float ff = 1.f / (1.f + expf(-g[DDIM + d]));
  float gg = tanhf(g[2 * DDIM + d]);
  float oo = 1.f / (1.f + expf(-g[3 * DDIM + d]));
  float cn = ff * cS[b * DDIM + d] + ii * gg;
  cS[b * DDIM + d] = cn;
  hS[b * DDIM + d] = oo * tanhf(cn);
}

__global__ __launch_bounds__(256) void k_att_init(float* rS, unsigned* gmax, float* gsum) {
  int b = blockIdx.x, d = threadIdx.x;
  rS[b * DDIM + d] = 0.f;
  if (d == 0) { gmax[b] = 0u; gsum[b] = 0.f; }
}

// e[n] = out[n] . h[batch[n]] ; segment max via encoded atomicMax (1 wave / node)
__global__ __launch_bounds__(256) void k_att_e(
    const float* __restrict__ outf, const float* __restrict__ hS,
    const int* __restrict__ batch, float* __restrict__ eN, unsigned* __restrict__ gmax) {
  int lane = threadIdx.x & 31;
  int n = blockIdx.x * 8 + (threadIdx.x >> 5);
  int g = batch[n];
  const float* orow = outf + (long)n * DDIM;
  const float* hrow = hS + (long)g * DDIM;
  float s = 0.f;
#pragma unroll
  for (int i = 0; i < 8; i++) s = fmaf(orow[lane + 32 * i], hrow[lane + 32 * i], s);
  for (int off = 16; off; off >>= 1) s += __shfl_xor(s, off, 32);
  if (lane == 0) { eN[n] = s; atomicMax(gmax + g, f_enc(s)); }
}

__global__ __launch_bounds__(256) void k_att_ex(
    const float* __restrict__ eN, const int* __restrict__ batch,
    const unsigned* __restrict__ gmax, float* __restrict__ exN,
    float* __restrict__ gsum) {
  int n = blockIdx.x * 256 + threadIdx.x;
  if (n >= NNODE) return;
  int g = batch[n];
  float e = expf(eN[n] - f_dec(gmax[g]));
  exN[n] = e;
  atomicAdd(gsum + g, e);
}

__global__ __launch_bounds__(256) void k_att_r(
    const float* __restrict__ exN, const float* __restrict__ gsum,
    const int* __restrict__ batch, const float* __restrict__ outf,
    float* __restrict__ rS) {
  int n = blockIdx.x, d = threadIdx.x;
  int g = batch[n];
  float a = exN[n] / gsum[g];
  atomicAdd(rS + (long)g * DDIM + d, a * outf[(long)n * DDIM + d]);
}

// ---------- readout: y = relu([h||r]@W1+b1)@W2+b2 ----------
__global__ __launch_bounds__(256) void k_readout(
    const float* __restrict__ hS, const float* __restrict__ rS,
    const float* __restrict__ W1, const float* __restrict__ b1,
    const float* __restrict__ W2, const float* __restrict__ b2,
    float* __restrict__ y) {
  __shared__ float red[256];
  int b = blockIdx.x, d = threadIdx.x;
  const float* hb = hS + (long)b * DDIM;
  const float* rb = rS + (long)b * DDIM;
  float s = b1[d];
  for (int k = 0; k < DDIM; k++) s = fmaf(hb[k], W1[k * DDIM + d], s);
  for (int k = 0; k < DDIM; k++) s = fmaf(rb[k], W1[(DDIM + k) * DDIM + d], s);
  s = fmaxf(s, 0.f);
  red[d] = s * W2[d];
  __syncthreads();
  for (int off = 128; off; off >>= 1) {
    if (d < off) red[d] += red[d + off];
    __syncthreads();
  }
  if (d == 0) y[b] = red[0] + b2[0];
}

extern "C" void kernel_launch(void* const* d_in, const int* in_sizes, int n_in,
                              void* d_out, int out_size, void* d_ws, size_t ws_size,
                              hipStream_t stream) {
  (void)in_sizes; (void)n_in; (void)out_size; (void)ws_size;
  const float* x   = (const float*)d_in[0];
  const float* ea  = (const float*)d_in[1];
  const float* W0  = (const float*)d_in[2];
  const float* b0  = (const float*)d_in[3];
  const float* Wm1 = (const float*)d_in[4];
  const float* bm1 = (const float*)d_in[5];
  const float* Wm2 = (const float*)d_in[6];
  const float* bm2 = (const float*)d_in[7];
  const float* Wr  = (const float*)d_in[8];
  const float* br  = (const float*)d_in[9];
  const float* Wih = (const float*)d_in[10];
  const float* Whh = (const float*)d_in[11];
  const float* bl  = (const float*)d_in[12];
  const float* W1  = (const float*)d_in[13];
  const float* b1  = (const float*)d_in[14];
  const float* W2  = (const float*)d_in[15];
  const float* b2  = (const float*)d_in[16];
  const int* ei    = (const int*)d_in[17];
  const int* batch = (const int*)d_in[18];
  const int* src = ei;
  const int* dst = ei + NEDGE;
  float* y = (float*)d_out;

  char* w = (char*)d_ws;
  auto take = [&](size_t bytes) -> char* {
    char* p = w; w += (bytes + 255) & ~(size_t)255; return p;
  };
  bf16_t* h0b  = (bf16_t*)take((size_t)NNODE * DDIM * 2);
  float*  agg  = (float*) take((size_t)NNODE * DDIM * 4);
  float*  outf = (float*) take((size_t)NNODE * DDIM * 4);
  bf16_t* Wm1T = (bf16_t*)take((size_t)DDIM * K1PAD * 2);
  bf16_t* Wm2T = (bf16_t*)take((size_t)DDIM * DDIM * 2);
  bf16_t* WrT  = (bf16_t*)take((size_t)DDIM * DDIM * 2);
  bf16_t* WgT  = (bf16_t*)take((size_t)4 * DDIM * KGATE * 2);
  bf16_t* qhb  = (bf16_t*)take((size_t)NGR * KGATE * 2);
  float*  gts  = (float*) take((size_t)NGR * 4 * DDIM * 4);
  float*  hS   = (float*) take((size_t)NGR * DDIM * 4);
  float*  cS   = (float*) take((size_t)NGR * DDIM * 4);
  float*  rS   = (float*) take((size_t)NGR * DDIM * 4);
  float*  eN   = (float*) take((size_t)NNODE * 4);
  float*  exN  = (float*) take((size_t)NNODE * 4);
  unsigned* gmax = (unsigned*)take((size_t)NGR * 4);
  float*  gsum = (float*) take((size_t)NGR * 4);

  const size_t lds_edges = (size_t)(DDIM * K1PAD + DDIM * DDIM + 4 * 16 * DDIM) * 2;
  const size_t lds_node  = (size_t)(DDIM * DDIM) * 2;

  k_prep<<<(4 * DDIM * KGATE + 255) / 256, 256, 0, stream>>>(Wm1, Wm2, Wr, Wih, Whh,
                                                             Wm1T, Wm2T, WrT, WgT);
  k_h0<<<NNODE, 256, 0, stream>>>(x, W0, b0, h0b, agg);
  k_edges<<<EDGE_BLOCKS, 128, lds_edges, stream>>>(h0b, ea, Wm1T, Wm2T, bm1, bm2,
                                                   src, dst, agg);
  k_node_out<<<NODE_BLOCKS, 128, lds_node, stream>>>(h0b, WrT, br, agg, outf);
  k_init_state<<<NGR * DDIM / 256, 256, 0, stream>>>(hS, cS, rS);

  for (int step = 0; step < 3; step++) {
    k_qh<<<NGR * KGATE / 256, 256, 0, stream>>>(hS, rS, qhb);
    k_gates<<<32, 128, 0, stream>>>(qhb, WgT, bl, gts);
    k_lstm<<<NGR, 256, 0, stream>>>(gts, cS, hS);
    k_att_init<<<NGR, 256, 0, stream>>>(rS, gmax, gsum);
    k_att_e<<<NNODE / 8, 256, 0, stream>>>(outf, hS, batch, eN, gmax);
    k_att_ex<<<(NNODE + 255) / 256, 256, 0, stream>>>(eN, batch, gmax, exN, gsum);
    k_att_r<<<NNODE, 256, 0, stream>>>(exN, gsum, batch, outf, rS);
  }
  k_readout<<<NGR, 256, 0, stream>>>(hS, rS, W1, b1, W2, b2, y);
}
